// LinearW2A8_50027779064442
// MI455X (gfx1250) — matrix operand order repaired and checked by reading, not compiled, not run on hardware
//
#include <hip/hip_runtime.h>
#include <stdint.h>

typedef __attribute__((ext_vector_type(16))) _Float16 v16h;
typedef __attribute__((ext_vector_type(8)))  _Float16 v8h;
typedef __attribute__((ext_vector_type(2)))  _Float16 h2;
typedef __attribute__((ext_vector_type(2)))  __fp16   cvt2;   // cvt_pkrtz return type
typedef __attribute__((ext_vector_type(8)))  float    v8f;

#define B_DIM    64
#define IN_DIM   8192
#define OUT_DIM  8192
#define GSZ      64
#define NGRP     (IN_DIM / GSZ)     // 128 groups per row
#define WROW_B   (IN_DIM / 4)       // 2048 packed bytes per W row
#define LDS_STR  72                 // halves per row: 144B, 16B-aligned, bank-spread
#define KSPLIT   4
#define KPB      (IN_DIM / KSPLIT)  // 2048 k per block

union AFrag { v16h v; v8h h[2]; };
union BFrag { v16h v; h2 p[8]; };
union Pack8 { v8h v; h2 p[4]; };

// ---------------------------------------------------------------------------
// Main GEMM: dst += x · dequant(W)^T for one (128-col, 2048-k) slab.
// K-permuted fragments: within each 16-k block, slot p holds k = p/2 + 8*(p&1);
// applied identically to A (at LDS staging) and B (unpack), so the WMMA
// contraction is order-invariant-correct while B unpack is 2 ops/pair.
// ---------------------------------------------------------------------------
__global__ __launch_bounds__(256) void linear_w2_wmma_kernel(
    const float*   __restrict__ x,       // [64, 8192] f32
    const uint8_t* __restrict__ wq,      // [8192, 2048] packed 2-bit LSB-first
    const float*   __restrict__ scales,  // [8192, 128]
    const int*     __restrict__ zps,     // [8192, 128]
    float*         __restrict__ dst,     // ws partials or final out
    int atomic_mode)
{
    __shared__ __attribute__((aligned(16))) uint16_t lds[2][64][LDS_STR];

    const int tid   = threadIdx.x;
    const int lane  = tid & 31;
    const int wid   = tid >> 5;          // 8 waves = 8 unique 16-wide N strips
    const int selhi = lane >> 4;
    const int lml   = lane & 15;

    const int n    = blockIdx.x * 128 + wid * 16 + lml;  // output column
    const int kz   = blockIdx.y;
    const int kbeg = kz * KPB;

    // cooperative staging indices: thread -> (row, 16-col block) of X chunk
    const int sr  = tid >> 2;
    const int scg = (tid & 3) * 16;

    const uint8_t* wrow = wq + (size_t)n * WROW_B;
    const float*   srow = scales + (size_t)n * NGRP;
    const int*     zrow = zps + (size_t)n * NGRP;

    v8f acc[4] = {{}, {}, {}, {}};

    for (int kc = kbeg, ci = 0; kc < kbeg + KPB; kc += GSZ, ++ci) {
        const int buf = ci & 1;
        const int g   = kc >> 6;

        // ---- per-chunk global fetches (64 codes + group qparams per lane) ----
        uint4 wbits = *(const uint4*)(wrow + (kc >> 2));
        float sc_f  = srow[g];
        int   zp_i  = zrow[g];

        // ---- stage X[0:64, kc:kc+64] f32 -> f16 into LDS, K-interleaved ----
        {
            const float4* xs = (const float4*)(x + (size_t)sr * IN_DIM + kc + scg);
            float4 f0 = xs[0], f1 = xs[1], f2 = xs[2], f3 = xs[3];
            float fl[16] = { f0.x, f0.y, f0.z, f0.w, f1.x, f1.y, f1.z, f1.w,
                             f2.x, f2.y, f2.z, f2.w, f3.x, f3.y, f3.z, f3.w };
            Pack8 ha, hb;
            #pragma unroll
            for (int d = 0; d < 4; ++d)       // slots 2d,2d+1 <- k = d, d+8
                ha.p[d] = __builtin_bit_cast(
                    h2, __builtin_amdgcn_cvt_pkrtz(fl[d], fl[d + 8]));
            #pragma unroll
            for (int d = 0; d < 4; ++d)       // slots 8+2d,9+2d <- k = 4+d, 12+d
                hb.p[d] = __builtin_bit_cast(
                    h2, __builtin_amdgcn_cvt_pkrtz(fl[4 + d], fl[12 + d]));
            uint16_t* p = &lds[buf][sr][scg];
            *(v8h*)(void*)p       = ha.v;
            *(v8h*)(void*)(p + 8) = hb.v;
        }

        __syncthreads();   // one barrier per chunk (double-buffered LDS)

        // dequant constants: ((1024+c) + (-(1024+zp))) * s, exact pk_add addend
        const _Float16 s_h = (_Float16)sc_f;
        const _Float16 a_h = (_Float16)(-(1024.0f + (float)zp_i));
        const h2 s2 = { s_h, s_h };
        const h2 a2 = { a_h, a_h };

        #pragma unroll
        for (int ks = 0; ks < 2; ++ks) {
            // B fragment: element pair (2j,2j+1) <- codes (j, j+8) of this word
            const uint32_t w32 = ((const uint32_t*)&wbits)[ks * 2 + selhi];
            BFrag b;
            #pragma unroll
            for (int j = 0; j < 8; ++j) {
                uint32_t p = ((w32 >> (2 * j)) & 0x00030003u) | 0x64006400u;
                h2 hp; __builtin_memcpy(&hp, &p, 4);
                b.p[j] = (hp + a2) * s2;       // v_pk_add_f16 + v_pk_mul_f16
            }

            // A fragments for 4 M tiles (permutation already applied in LDS)
            const int ofs = ks * 32 + selhi * 16;
            #pragma unroll
            for (int t = 0; t < 4; ++t) {
                AFrag a;
                a.h[0] = *(const v8h*)(const void*)&lds[buf][t * 16 + lml][ofs];
                a.h[1] = *(const v8h*)(const void*)&lds[buf][t * 16 + lml][ofs + 8];
                acc[t] = __builtin_amdgcn_wmma_f32_16x16x32_f16(
                             false, a.v, false, b.v, (short)0, acc[t], false, false);
            }
        }
    }

    // ---- epilogue: C layout VGPR r -> M = t*16 + r + 8*selhi, N = lane&15 ----
    const int mro = selhi * 8;
    if (atomic_mode) {
        #pragma unroll
        for (int t = 0; t < 4; ++t)
            #pragma unroll
            for (int r = 0; r < 8; ++r)
                atomicAdd(&dst[(size_t)(t * 16 + mro + r) * OUT_DIM + n], acc[t][r]);
    } else {
        float* base = dst + (size_t)kz * B_DIM * OUT_DIM;
        #pragma unroll
        for (int t = 0; t < 4; ++t)
            #pragma unroll
            for (int r = 0; r < 8; ++r)
                base[(size_t)(t * 16 + mro + r) * OUT_DIM + n] = acc[t][r];
    }
}

// out[m,n] = bias[n]  (atomic-mode pre-init)
__global__ __launch_bounds__(256) void init_bias_kernel(
    const float* __restrict__ bias, float* __restrict__ out)
{
    int e   = (blockIdx.x * 256 + threadIdx.x) * 4;
    int col = e & (OUT_DIM - 1);
    *(float4*)(out + e) = *(const float4*)(bias + col);
}

// out = bias + sum of KSPLIT partial slabs  (deterministic reduction)
__global__ __launch_bounds__(256) void reduce_kernel(
    const float* __restrict__ part, const float* __restrict__ bias,
    float* __restrict__ out)
{
    int e   = (blockIdx.x * 256 + threadIdx.x) * 4;
    int col = e & (OUT_DIM - 1);
    float4 s = *(const float4*)(bias + col);
    #pragma unroll
    for (int z = 0; z < KSPLIT; ++z) {
        float4 p = *(const float4*)(part + (size_t)z * B_DIM * OUT_DIM + e);
        s.x += p.x; s.y += p.y; s.z += p.z; s.w += p.w;
    }
    *(float4*)(out + e) = s;
}

extern "C" void kernel_launch(void* const* d_in, const int* in_sizes, int n_in,
                              void* d_out, int out_size, void* d_ws, size_t ws_size,
                              hipStream_t stream) {
    const float*   x      = (const float*)d_in[0];
    const uint8_t* wq     = (const uint8_t*)d_in[1];
    const float*   scales = (const float*)d_in[2];
    const int*     zps    = (const int*)d_in[3];
    const float*   bias   = (const float*)d_in[4];
    float*         out    = (float*)d_out;

    const dim3 grid(OUT_DIM / 128, KSPLIT);   // 64 x 4 = 256 workgroups
    const dim3 block(256);                    // 8 waves (wave32)
    const int  nvec = (B_DIM * OUT_DIM) / 4 / 256;  // 512 blocks for epilogues

    const size_t ws_need = (size_t)KSPLIT * B_DIM * OUT_DIM * sizeof(float); // 8 MB
    if (ws_size >= ws_need) {
        hipLaunchKernelGGL(linear_w2_wmma_kernel, grid, block, 0, stream,
                           x, wq, scales, zps, (float*)d_ws, 0);
        hipLaunchKernelGGL(reduce_kernel, dim3(nvec), block, 0, stream,
                           (const float*)d_ws, bias, out);
    } else {
        hipLaunchKernelGGL(init_bias_kernel, dim3(nvec), block, 0, stream, bias, out);
        hipLaunchKernelGGL(linear_w2_wmma_kernel, grid, block, 0, stream,
                           x, wq, scales, zps, out, 1);
    }
}
